// MoE_42021960024675
// MI455X (gfx1250) — matrix operand order repaired and checked by reading, not compiled, not run on hardware
//
#include <hip/hip_runtime.h>
#include <hip/hip_bf16.h>

// ---------------------------------------------------------------------------
// MoE forward for MI455X (gfx1250, wave32, WMMA bf16).
//   B=4, S=2048 -> T=8192 tokens, D=1024, F=4096, E=8, top-2 routing.
// Pass 0 (if workspace allows): repack f32 weights -> bf16 (one sweep).
// Pass 1: router (logits -> softmax -> top2 -> normalized combine weights).
// Pass 2: fused expert GEMMs, 32-token M-tile, v_wmma_f32_16x16x32_bf16.
// ---------------------------------------------------------------------------

#define D_DIM 1024
#define F_DIM 4096
#define E_NUM 8
#define FC    256          // F-chunk per block iteration (16 h-cols per wave)
#define XPAD  8            // row padding (elements) to break LDS bank conflicts
#define HPAD  8

typedef __attribute__((ext_vector_type(16))) __bf16 v16bf;
typedef __attribute__((ext_vector_type(8)))  __bf16 v8bf;
typedef __attribute__((ext_vector_type(16))) float  v16f;
typedef __attribute__((ext_vector_type(8)))  float  v8fv;
typedef __attribute__((ext_vector_type(8)))  float  v8f;

union AFrag { v16bf v; uint4 q[2]; };
union BSrc  { v16f  v; float4 q[4]; };

static __device__ inline v8f zero8() {
    v8f z;
    #pragma unroll
    for (int i = 0; i < 8; ++i) z[i] = 0.0f;
    return z;
}

// A-fragment (16x32 bf16, lane = M 0..15, half = lane/16):
//   elems 0..15 -> K = k0 + half*16 + j
// Both halves 16B-contiguous in LDS -> two ds_load_b128, no packing VALU.
static __device__ inline v16bf load_a_frag(const __bf16* row, int k0, int half) {
    AFrag f;
    f.q[0] = *(const uint4*)(row + k0 + half * 16);
    f.q[1] = *(const uint4*)(row + k0 + half * 16 + 8);
    return f.v;
}

// B-fragment from f32 weights: 4x global_load_b128 + packed v_cvt_pk_bf16_f32.
static __device__ inline v16bf load_b_frag(const float* p) {
    BSrc s;
    s.q[0] = ((const float4*)p)[0];
    s.q[1] = ((const float4*)p)[1];
    s.q[2] = ((const float4*)p)[2];
    s.q[3] = ((const float4*)p)[3];
    return __builtin_convertvector(s.v, v16bf);
}

// B-fragment from repacked bf16 weights: 2x global_load_b128, zero cvt.
static __device__ inline v16bf load_b_frag(const __bf16* p) {
    AFrag f;
    f.q[0] = ((const uint4*)p)[0];
    f.q[1] = ((const uint4*)p)[1];
    return f.v;
}

static __device__ inline v8f wmma_bf16(v16bf a, v16bf b, v8f c) {
    return __builtin_amdgcn_wmma_f32_16x16x32_bf16(
        /*neg_a=*/false, a, /*neg_b=*/false, b,
        /*c_mod=*/(short)0, c, /*reuse_a=*/false, /*reuse_b=*/false);
}

// sigmoid via hardware v_rcp_f32 (no IEEE division sequence)
static __device__ inline float fast_sigmoid(float g) {
    return __builtin_amdgcn_rcpf(1.0f + __expf(-g));
}

// ---------------------------------------------------------------------------
// Pass 0: f32 -> bf16 weight repack (8 elements per thread, b128 in/out).
// ---------------------------------------------------------------------------
__global__ __launch_bounds__(256, 1)
void cvt_f32_bf16_kernel(const float* __restrict__ src,
                         __bf16* __restrict__ dst, size_t n) {
    const size_t i = ((size_t)blockIdx.x * 256 + threadIdx.x) * 8;
    if (i >= n) return;
    union { v8fv v; float4 q[2]; } s;
    s.q[0] = ((const float4*)(src + i))[0];
    s.q[1] = ((const float4*)(src + i))[1];
    *(v8bf*)(dst + i) = __builtin_convertvector(s.v, v8bf);
}

// ---------------------------------------------------------------------------
// Pass 1: router. One wave per token; block = 8 waves = 8 tokens.
// ---------------------------------------------------------------------------
__global__ __launch_bounds__(256, 1)
void moe_router_kernel(const float* __restrict__ x,
                       const float* __restrict__ Wg,
                       float* __restrict__ comb, int T) {
    const int lane = threadIdx.x & 31;
    const int wave = threadIdx.x >> 5;
    const int t = blockIdx.x * 8 + wave;
    if (t >= T) return;

    float a[E_NUM];
    #pragma unroll
    for (int e = 0; e < E_NUM; ++e) a[e] = 0.0f;

    const float* xr = x + (size_t)t * D_DIM;
    for (int d = lane; d < D_DIM; d += 32) {
        const float xv = xr[d];
        #pragma unroll
        for (int e = 0; e < E_NUM; ++e) a[e] += xv * Wg[e * D_DIM + d];
    }
    #pragma unroll
    for (int e = 0; e < E_NUM; ++e) {
        #pragma unroll
        for (int off = 16; off > 0; off >>= 1)
            a[e] += __shfl_xor(a[e], off, 32);
    }
    if (lane == 0) {
        float mx = a[0];
        #pragma unroll
        for (int e = 1; e < E_NUM; ++e) mx = fmaxf(mx, a[e]);
        float p[E_NUM];
        #pragma unroll
        for (int e = 0; e < E_NUM; ++e) p[e] = __expf(a[e] - mx);
        int i1 = 0;
        #pragma unroll
        for (int e = 1; e < E_NUM; ++e) if (p[e] > p[i1]) i1 = e;
        int i2 = (i1 == 0) ? 1 : 0;
        #pragma unroll
        for (int e = 0; e < E_NUM; ++e)
            if (e != i1 && p[e] > p[i2]) i2 = e;
        const float inv = 1.0f / (p[i1] + p[i2]);
        #pragma unroll
        for (int e = 0; e < E_NUM; ++e)
            comb[(size_t)t * E_NUM + e] = (e == i1 || e == i2) ? p[e] * inv : 0.0f;
    }
}

// ---------------------------------------------------------------------------
// Pass 2: fused expert GEMMs. Block = 32 tokens x 16 waves (512 threads).
// Every B fragment is reused across 2 M-tiles (a0/a1), halving weight traffic.
// Wave w: gate/up -> h-columns [fc+w*16, fc+w*16+16) for all 32 tokens;
//         down    -> output columns [w*64, w*64+64) for all 32 tokens.
// Routing weight is folded into h, so down-projection accumulates the final
// output directly: no atomics, no output pre-zeroing.
// ---------------------------------------------------------------------------
template <typename WT>
__global__ __launch_bounds__(512, 1)
void moe_expert_kernel(const float* __restrict__ x,
                       const WT* __restrict__ wg,
                       const WT* __restrict__ wu,
                       const WT* __restrict__ wd,
                       const float* __restrict__ comb,
                       float* __restrict__ out) {
    __shared__ __bf16 xs[32][D_DIM + XPAD];   // ~66 KB
    __shared__ __bf16 hs[32][FC + HPAD];      // ~17 KB
    __shared__ float  combs[32][E_NUM];       // 1 KB

    const int tid  = threadIdx.x;
    const int lane = tid & 31;
    const int wave = tid >> 5;                // 0..15
    const int m0   = blockIdx.x * 32;         // token tile base

    // Stage x tile -> bf16 LDS (hardware cvt).
    for (int i = tid; i < 32 * D_DIM; i += 512) {
        const int m = i >> 10;
        const int k = i & (D_DIM - 1);
        xs[m][k] = (__bf16)x[(size_t)(m0 + m) * D_DIM + k];
    }
    if (tid < 32 * E_NUM) {
        const int m = tid >> 3, e = tid & 7;
        combs[m][e] = comb[(size_t)(m0 + m) * E_NUM + e];
    }
    __syncthreads();

    const int mrow = lane & 15;               // M within tile / N for B-frags
    const int half = lane >> 4;

    v8f acc[2][4];                            // [M-tile][N-tile], 64 VGPRs
    #pragma unroll
    for (int mt = 0; mt < 2; ++mt)
        #pragma unroll
        for (int t = 0; t < 4; ++t) acc[mt][t] = zero8();

    for (int e = 0; e < E_NUM; ++e) {
        // Tile-uniform skip: expert unused by all 32 tokens contributes zero.
        bool any = false;
        #pragma unroll
        for (int m = 0; m < 32; ++m) any |= (combs[m][e] != 0.0f);
        if (!any) continue;

        const WT* wge = wg + (size_t)e * F_DIM * D_DIM;
        const WT* wue = wu + (size_t)e * F_DIM * D_DIM;
        const WT* wde = wd + (size_t)e * D_DIM * F_DIM;

        for (int fc = 0; fc < F_DIM; fc += FC) {
            const int fw = fc + wave * 16;    // this wave's 16 h-columns
            v8f g0 = zero8(), g1 = zero8(), u0 = zero8(), u1 = zero8();

            for (int k0 = 0; k0 < D_DIM; k0 += 32) {
                const v16bf a0 = load_a_frag(xs[mrow], k0, half);
                const v16bf a1 = load_a_frag(xs[16 + mrow], k0, half);
                const size_t r = (size_t)(fw + mrow) * D_DIM + k0 + half * 16;
                // Speculative next-K prefetch (branch-free; OOB is dropped).
                __builtin_prefetch(wge + r + 32, 0, 1);
                __builtin_prefetch(wue + r + 32, 0, 1);
                const v16bf bg = load_b_frag(wge + r);
                const v16bf bu = load_b_frag(wue + r);
                g0 = wmma_bf16(a0, bg, g0);
                g1 = wmma_bf16(a1, bg, g1);   // B reused across M-tiles
                u0 = wmma_bf16(a0, bu, u0);
                u1 = wmma_bf16(a1, bu, u1);
            }

            // h = comb * sigmoid(gate) * up -> bf16 LDS (routing weight folded).
            #pragma unroll
            for (int v = 0; v < 8; ++v) {
                const int ma = v + half * 8;        // C-layout: M = v + 8*half
                const int mb = 16 + ma;
                const float ca = combs[ma][e];
                const float cb = combs[mb][e];
                hs[ma][wave * 16 + mrow] = (__bf16)(ca * u0[v] * fast_sigmoid(g0[v]));
                hs[mb][wave * 16 + mrow] = (__bf16)(cb * u1[v] * fast_sigmoid(g1[v]));
            }
            __syncthreads();

            // Down-projection: acc += h[32 x 256] @ w_down[dcols, fc:fc+256]^T
            for (int kf = 0; kf < FC; kf += 32) {
                const v16bf ah0 = load_a_frag(hs[mrow], kf, half);
                const v16bf ah1 = load_a_frag(hs[16 + mrow], kf, half);
                #pragma unroll
                for (int t = 0; t < 4; ++t) {
                    const int dcol = wave * 64 + t * 16 + mrow;
                    const v16bf bd =
                        load_b_frag(wde + (size_t)dcol * F_DIM + fc + kf + half * 16);
                    acc[0][t] = wmma_bf16(ah0, bd, acc[0][t]);
                    acc[1][t] = wmma_bf16(ah1, bd, acc[1][t]);
                }
            }
            __syncthreads();                  // before hs is overwritten
        }
    }

    // Write back (each block/wave owns a disjoint [32 x 64] slab).
    #pragma unroll
    for (int mt = 0; mt < 2; ++mt)
        #pragma unroll
        for (int t = 0; t < 4; ++t) {
            const int d = wave * 64 + t * 16 + mrow;
            #pragma unroll
            for (int v = 0; v < 8; ++v) {
                const int m = mt * 16 + v + half * 8;
                out[(size_t)(m0 + m) * D_DIM + d] = acc[mt][t][v];
            }
        }
}

extern "C" void kernel_launch(void* const* d_in, const int* in_sizes, int n_in,
                              void* d_out, int out_size, void* d_ws, size_t ws_size,
                              hipStream_t stream) {
    const float* x     = (const float*)d_in[0];   // [B,S,D] f32
    const float* Wg    = (const float*)d_in[1];   // [E,D]
    const float* wgate = (const float*)d_in[2];   // [E,F,D]
    const float* wup   = (const float*)d_in[3];   // [E,F,D]
    const float* wdown = (const float*)d_in[4];   // [E,D,F]
    float* out = (float*)d_out;

    const int T = in_sizes[0] / D_DIM;            // 8192 tokens
    const size_t WELEM  = (size_t)E_NUM * F_DIM * D_DIM;   // 33.5M per matrix
    const size_t WBYTES = WELEM * sizeof(__bf16);
    const size_t NEED   = 3 * WBYTES + (size_t)T * E_NUM * sizeof(float);

    if (ws_size >= NEED) {
        // Fast path: one-shot bf16 weight repack, then bf16-weight GEMMs.
        __bf16* wgb = (__bf16*)d_ws;
        __bf16* wub = wgb + WELEM;
        __bf16* wdb = wub + WELEM;
        float*  comb = (float*)((char*)d_ws + 3 * WBYTES);

        const int cblocks = (int)((WELEM / 8 + 255) / 256);
        cvt_f32_bf16_kernel<<<cblocks, 256, 0, stream>>>(wgate, wgb, WELEM);
        cvt_f32_bf16_kernel<<<cblocks, 256, 0, stream>>>(wup,   wub, WELEM);
        cvt_f32_bf16_kernel<<<cblocks, 256, 0, stream>>>(wdown, wdb, WELEM);
        moe_router_kernel<<<(T + 7) / 8, 256, 0, stream>>>(x, Wg, comb, T);
        moe_expert_kernel<__bf16><<<T / 32, 512, 0, stream>>>(
            x, wgb, wub, wdb, comb, out);
    } else {
        // Fallback: stream f32 weights, convert per fragment.
        float* comb = (float*)d_ws;
        moe_router_kernel<<<(T + 7) / 8, 256, 0, stream>>>(x, Wg, comb, T);
        moe_expert_kernel<float><<<T / 32, 512, 0, stream>>>(
            x, wgate, wup, wdown, comb, out);
    }
}
